// OlmoAttention_28406913696241
// MI455X (gfx1250) — matrix operand order repaired and checked by reading, not compiled, not run on hardware
//
#include <hip/hip_runtime.h>

// ---------------------------------------------------------------------------
// Types for CDNA5 WMMA (wave32): v_wmma_f32_16x16x32_bf16
// ---------------------------------------------------------------------------
typedef __bf16 bf16;
typedef __bf16 v16bf __attribute__((ext_vector_type(16)));
typedef __bf16 v8bf  __attribute__((ext_vector_type(8)));
typedef float  v8f   __attribute__((ext_vector_type(8)));
typedef unsigned int u32x4 __attribute__((ext_vector_type(4)));
typedef int          i32x4 __attribute__((ext_vector_type(4)));
typedef int          i32x8 __attribute__((ext_vector_type(8)));

union Frag16 { v16bf v; v8bf h[2]; };

__device__ __forceinline__ v8f wmma_bf16(v16bf a, v16bf b, v8f c) {
  // 8 args: (neg_a, A, neg_b, B, c_mod, C, reuse_a, reuse_b)
  return __builtin_amdgcn_wmma_f32_16x16x32_bf16(false, a, false, b, (short)0, c,
                                                 false, false);
}

#define H_DIM    4096
#define NHEADS   32
#define HEAD_D   128
#define BATCH    2
#define SEQ      2048
#define M_TOT    (BATCH * SEQ)          // 4096
#define N_QKV    (3 * H_DIM)            // 12288
#define CLIP_V   8.0f
#define SM_SCALE 0.08838834764831845f   // 128^-0.5
#define LN_THETA 9.210340371976184f     // ln(10000)

// ---------------------------------------------------------------------------
// Tensor Data Mover: build a 2D-tile D# and issue TENSOR_LOAD_TO_LDS.
// Tile rows of tile_d0 bf16 elements, tile_d1 rows, row stride stride_elems.
// LDS destination gets pad_amount dwords appended every pad_interval dwords
// (hardware generates the bank-conflict-free padded layout for us).
//   pad_interval_code: dwords-before-pad = 2 << code
//   pad_amount_code:   pad dwords = code + 1
// ---------------------------------------------------------------------------
__device__ __forceinline__ unsigned ldsoff(const void* p) {
  // flat LDS pointer truncates to the 32-bit LDS byte offset (ISA 10.2)
  return (unsigned)(unsigned long long)p;
}

__device__ __forceinline__ void tdm_load_2d(const void* gsrc, unsigned lds_off,
                                            unsigned tile_d0, unsigned tile_d1,
                                            unsigned long long stride_elems,
                                            unsigned pad_interval_code,
                                            unsigned pad_amount_code) {
  const unsigned long long ga = (unsigned long long)gsrc;
  u32x4 g0;
  g0[0] = 1u;                                   // count=1 (valid user D#)
  g0[1] = lds_off;                              // D#.lds_addr (bytes)
  g0[2] = (unsigned)ga;                         // global_addr[31:0]
  g0[3] = (unsigned)((ga >> 32) & 0x01ffffffu)  // global_addr[56:32]
          | (2u << 30);                         // type = 2 ("image")
  i32x8 g1;
  g1[0] = (int)((1u << 16)                      // data_size = 2 bytes (bf16)
                | (1u << 20)                    // pad_enable
                | (pad_interval_code << 22) | (pad_amount_code << 25));
  // tensor_dim0[79:48] = tile_d0, tensor_dim1[111:80] = tile_d1 (no OOB),
  // tile_dim0[127:112], tile_dim1[143:128], tile_dim2 = 0
  g1[1] = (int)((tile_d0 & 0xffffu) << 16);
  g1[2] = (int)((tile_d0 >> 16) | ((tile_d1 & 0xffffu) << 16));
  g1[3] = (int)((tile_d1 >> 16) | (tile_d0 << 16));
  g1[4] = (int)(tile_d1 & 0xffffu);
  // tensor_dim0_stride[207:160] (elems), tensor_dim1_stride unused (2D)
  g1[5] = (int)(unsigned)(stride_elems & 0xffffffffu);
  g1[6] = (int)(unsigned)((stride_elems >> 32) & 0xffffu);
  g1[7] = 0;
  const i32x4 z4 = {0, 0, 0, 0};
#if __clang_major__ >= 23
  const i32x8 z8 = {0, 0, 0, 0, 0, 0, 0, 0};
  __builtin_amdgcn_tensor_load_to_lds(g0, g1, z4, z4, z8, 0);
#else
  __builtin_amdgcn_tensor_load_to_lds(g0, g1, z4, z4, 0);
#endif
}

// ---------------------------------------------------------------------------
// Kernel 1a: fp32 -> bf16 bulk convert (vectorized)
// ---------------------------------------------------------------------------
struct __align__(8) bf16x4 { bf16 a, b, c, d; };

__global__ __launch_bounds__(256) void cvt_f32_bf16_k(
    const float4* __restrict__ in, bf16x4* __restrict__ out, int n4) {
  int i = blockIdx.x * blockDim.x + threadIdx.x;
  if (i < n4) {
    float4 f = in[i];
    bf16x4 o = { (bf16)f.x, (bf16)f.y, (bf16)f.z, (bf16)f.w };
    out[i] = o;
  }
}

// ---------------------------------------------------------------------------
// Kernel 1b: transpose + convert:  W (R x C) fp32 -> WT (C x R) bf16
// ---------------------------------------------------------------------------
__global__ __launch_bounds__(256) void transpose_cvt_k(
    const float* __restrict__ W, bf16* __restrict__ WT, int R, int C) {
  __shared__ float t[32][33];
  const int gx = blockIdx.x * 32;   // source column
  const int gy = blockIdx.y * 32;   // source row
  const int tx = threadIdx.x, ty = threadIdx.y;
#pragma unroll
  for (int i = ty; i < 32; i += 8)
    t[i][tx] = W[(size_t)(gy + i) * C + gx + tx];
  __syncthreads();
#pragma unroll
  for (int i = ty; i < 32; i += 8)
    WT[(size_t)(gx + i) * R + gy + tx] = (bf16)t[tx][i];
}

// ---------------------------------------------------------------------------
// Kernel 2/5: TDM-fed double-buffered block GEMM
//   C(MxN) = A(MxK, row-major bf16) * B(KxN, row-major bf16)
// 256 threads = 8 waves; block tile 128x128; wave tile 64x32 (4x2 WMMA tiles);
// K-step 64 (2 WMMA k-chunks per staged tile).
// Wave 0 posts TENSOR_LOAD_TO_LDS descriptors for the next A/B tiles while
// all waves compute on the current buffer; completion via s_wait_tensorcnt.
// TDM pad fields generate LDS strides 72 (A) / 136 (B) -> no bank conflicts.
// EPI==1: clip +-8, store bf16.  EPI==0: store f32.
// ---------------------------------------------------------------------------
template <int EPI>
__global__ __launch_bounds__(256) void gemm_bf16_k(
    const bf16* __restrict__ A, const bf16* __restrict__ B,
    void* __restrict__ C, int M, int N, int K) {
  constexpr int KT = 64;
  constexpr int AS = 72;    // 64 elem row + 8 pad  (TDM: 32 dw interval, 4 dw pad)
  constexpr int BS = 136;   // 128 elem row + 8 pad (TDM: 64 dw interval, 4 dw pad)
  __shared__ __align__(32) bf16 As[2][128 * AS];
  __shared__ __align__(32) bf16 Bs[2][64 * BS];

  const int tid  = threadIdx.x;
  const int lane = tid & 31;
  const int wv   = tid >> 5;        // 0..7
  const int waveM = (wv >> 2) * 64;
  const int waveN = (wv & 3) * 32;
  const int m0 = blockIdx.y * 128;
  const int n0 = blockIdx.x * 128;
  const int rA  = lane & 15;
  const int hi  = lane >> 4;
  const int kb8 = hi * 8;

  v8f acc[4][2];
#pragma unroll
  for (int i = 0; i < 4; ++i)
#pragma unroll
    for (int j = 0; j < 2; ++j) acc[i][j] = v8f{0.f,0.f,0.f,0.f,0.f,0.f,0.f,0.f};

  const int niter = K / KT;

  // ---- prologue: stage tile 0 via TDM ----
  if (wv == 0) {
    tdm_load_2d(A + (size_t)m0 * K, ldsoff(&As[0][0]), KT, 128,
                (unsigned long long)K, /*32 dw*/4, /*4 dw*/3);
    tdm_load_2d(B + (size_t)n0, ldsoff(&Bs[0][0]), 128, KT,
                (unsigned long long)N, /*64 dw*/5, /*4 dw*/3);
    __builtin_amdgcn_s_wait_tensorcnt(0);
  }
  __syncthreads();

  for (int it = 0; it < niter; ++it) {
    const int cur = it & 1;
    // ---- post DMA for next tile into the other buffer ----
    if (wv == 0 && it + 1 < niter) {
      const int nxt = cur ^ 1;
      tdm_load_2d(A + (size_t)m0 * K + (size_t)(it + 1) * KT,
                  ldsoff(&As[nxt][0]), KT, 128, (unsigned long long)K, 4, 3);
      tdm_load_2d(B + (size_t)(it + 1) * KT * N + n0,
                  ldsoff(&Bs[nxt][0]), 128, KT, (unsigned long long)N, 5, 3);
    }
    const bf16* as = &As[cur][0];
    const bf16* bs = &Bs[cur][0];
#pragma unroll
    for (int kk = 0; kk < 2; ++kk) {
      // A frags: row = lane&15, K elems {kb8..+7, 16+kb8..+7} within chunk
      Frag16 af[4];
#pragma unroll
      for (int mi = 0; mi < 4; ++mi) {
        const int r = waveM + mi * 16 + rA;
        af[mi].h[0] = *(const v8bf*)&as[r * AS + kk * 32 + kb8];
        af[mi].h[1] = *(const v8bf*)&as[r * AS + kk * 32 + kb8 + 16];
      }
      // B frags: col = lane&15, K rows {kb8..+7, 16+kb8..+7} within chunk
#pragma unroll
      for (int ni = 0; ni < 2; ++ni) {
        Frag16 bfr;
#pragma unroll
        for (int s = 0; s < 16; ++s)
          bfr.v[s] = bs[(kk * 32 + (s < 8 ? kb8 + s : 8 + kb8 + s)) * BS + waveN + ni * 16 + rA];
#pragma unroll
        for (int mi = 0; mi < 4; ++mi)
          acc[mi][ni] = wmma_bf16(af[mi].v, bfr.v, acc[mi][ni]);
      }
    }
    if (wv == 0) __builtin_amdgcn_s_wait_tensorcnt(0);
    __syncthreads();
  }

  // ---- epilogue: C layout row = r + 8*hi, col = lane&15 ----
#pragma unroll
  for (int mi = 0; mi < 4; ++mi)
#pragma unroll
    for (int ni = 0; ni < 2; ++ni)
#pragma unroll
      for (int r = 0; r < 8; ++r) {
        const int grow = m0 + waveM + mi * 16 + r + 8 * hi;
        const int gcol = n0 + waveN + ni * 16 + rA;
        float x = acc[mi][ni][r];
        if constexpr (EPI == 1) {
          x = fminf(fmaxf(x, -CLIP_V), CLIP_V);
          ((bf16*)C)[(size_t)grow * N + gcol] = (bf16)x;
        } else {
          ((float*)C)[(size_t)grow * N + gcol] = x;
        }
      }
}

// ---------------------------------------------------------------------------
// Kernel 3: RoPE + relayout.
//   qkv (M_TOT x 3H bf16, clipped)  ->
//     Q  [bh, s, d]   (A-matrix friendly)
//     KT [bh, d, s]   (B-matrix friendly for Q.K^T: lane=key, contiguous keys)
//     V  [bh, s, d]   (B-matrix friendly for P.V : lane=d, contiguous d)
// ---------------------------------------------------------------------------
__global__ __launch_bounds__(256) void rope_relayout_k(
    const bf16* __restrict__ qkv, const int* __restrict__ pos,
    bf16* __restrict__ Q, bf16* __restrict__ KT, bf16* __restrict__ V) {
  const int idx = blockIdx.x * blockDim.x + threadIdx.x;  // < B*S*NH*64 = 2^23
  const int d = idx & 63;
  const int h = (idx >> 6) & (NHEADS - 1);
  const int s = (idx >> 11) & (SEQ - 1);
  const int b = idx >> 22;

  const bf16* row = qkv + (size_t)(b * SEQ + s) * N_QKV;
  const float q1 = (float)row[h * HEAD_D + d];
  const float q2 = (float)row[h * HEAD_D + d + 64];
  const float k1 = (float)row[H_DIM + h * HEAD_D + d];
  const float k2 = (float)row[H_DIM + h * HEAD_D + d + 64];
  const bf16  v1 = row[2 * H_DIM + h * HEAD_D + d];
  const bf16  v2 = row[2 * H_DIM + h * HEAD_D + d + 64];

  const float p   = (float)pos[s];
  const float inv = __expf(-((float)d * (1.0f / 64.0f)) * LN_THETA);
  float c, sn;
  __sincosf(p * inv, &sn, &c);

  const size_t bh = (size_t)(b * NHEADS + h);
  Q[(bh * SEQ + s) * HEAD_D + d]        = (bf16)(q1 * c - q2 * sn);
  Q[(bh * SEQ + s) * HEAD_D + d + 64]   = (bf16)(q2 * c + q1 * sn);
  KT[(bh * HEAD_D + d) * SEQ + s]       = (bf16)(k1 * c - k2 * sn);
  KT[(bh * HEAD_D + d + 64) * SEQ + s]  = (bf16)(k2 * c + k1 * sn);
  V[(bh * SEQ + s) * HEAD_D + d]        = v1;
  V[(bh * SEQ + s) * HEAD_D + d + 64]   = v2;
}

// ---------------------------------------------------------------------------
// Kernel 4: causal flash attention (wave-level).
// Block = 128 threads = 4 waves; each wave owns 16 query rows.
// Q(16x128) kept in A-frags; per 64-key chunk: 16 WMMA (QK^T) + online
// softmax (shfl_xor row reductions) + P->LDS->A-frag + 16 WMMA (PV).
// ---------------------------------------------------------------------------
__global__ __launch_bounds__(128) void flash_attn_k(
    const bf16* __restrict__ Q, const bf16* __restrict__ KT,
    const bf16* __restrict__ V, bf16* __restrict__ attn) {
  constexpr int PS = 72;  // 64 + 8 pad
  __shared__ __align__(32) bf16 Plds[4][16 * PS];

  const int lane = threadIdx.x & 31;
  const int wv   = threadIdx.x >> 5;
  const int bh   = blockIdx.y;                    // b*NH + h
  const int q0   = (blockIdx.x * 4 + wv) * 16;    // query row base (per batch)
  const int rA   = lane & 15;
  const int hi   = lane >> 4;
  const int kb8  = hi * 8;

  const bf16* qb = Q + ((size_t)bh * SEQ + q0) * HEAD_D;
  Frag16 aq[4];
#pragma unroll
  for (int c = 0; c < 4; ++c) {
    aq[c].h[0] = *(const v8bf*)&qb[rA * HEAD_D + c * 32 + kb8];
    aq[c].h[1] = *(const v8bf*)&qb[rA * HEAD_D + c * 32 + kb8 + 16];
  }

  v8f o[8];
#pragma unroll
  for (int n = 0; n < 8; ++n) o[n] = v8f{0.f,0.f,0.f,0.f,0.f,0.f,0.f,0.f};
  float m[8], l[8];
#pragma unroll
  for (int r = 0; r < 8; ++r) { m[r] = -1e30f; l[r] = 0.f; }

  bf16* P = &Plds[wv][0];
  const int nchunk = q0 / 64 + 1;  // causal: only key chunks with base <= q0+15

  for (int j = 0; j < nchunk; ++j) {
    const int kbase = j * 64;

    // ---- scores S = Q * K^T : 4 N-tiles (16 keys) x 4 K-chunks ----
    v8f sc[4];
#pragma unroll
    for (int nt = 0; nt < 4; ++nt) sc[nt] = v8f{0.f,0.f,0.f,0.f,0.f,0.f,0.f,0.f};
#pragma unroll
    for (int nt = 0; nt < 4; ++nt)
#pragma unroll
      for (int c = 0; c < 4; ++c) {
        Frag16 bk;  // col = key (lane&15), elements = d in table order
        const bf16* kp =
            KT + ((size_t)bh * HEAD_D + c * 32) * SEQ + kbase + nt * 16 + rA;
#pragma unroll
        for (int s = 0; s < 16; ++s)
          bk.v[s] = kp[(size_t)(s < 8 ? kb8 + s : 8 + kb8 + s) * SEQ];
        sc[nt] = wmma_bf16(aq[c].v, bk.v, sc[nt]);
      }

    // ---- scale + causal mask ----
#pragma unroll
    for (int nt = 0; nt < 4; ++nt)
#pragma unroll
      for (int r = 0; r < 8; ++r) {
        const int key = kbase + nt * 16 + rA;
        const int qr  = q0 + r + 8 * hi;
        float x = sc[nt][r] * SM_SCALE;
        sc[nt][r] = (key > qr) ? -1e30f : x;
      }

    // ---- online softmax (row stats replicated across each 16-lane half) ----
#pragma unroll
    for (int r = 0; r < 8; ++r) {
      float mx = fmaxf(fmaxf(sc[0][r], sc[1][r]), fmaxf(sc[2][r], sc[3][r]));
#pragma unroll
      for (int off = 1; off < 16; off <<= 1) mx = fmaxf(mx, __shfl_xor(mx, off, 32));
      const float mn    = fmaxf(m[r], mx);
      const float alpha = __expf(m[r] - mn);
      float sum = 0.f;
#pragma unroll
      for (int nt = 0; nt < 4; ++nt) {
        const float pex = __expf(sc[nt][r] - mn);
        sc[nt][r] = pex;
        sum += pex;
      }
#pragma unroll
      for (int off = 1; off < 16; off <<= 1) sum += __shfl_xor(sum, off, 32);
      m[r] = mn;
      l[r] = l[r] * alpha + sum;
#pragma unroll
      for (int n = 0; n < 8; ++n) o[n][r] *= alpha;
    }

    // ---- P (C-layout) -> LDS -> A-layout frags (per-wave buffer) ----
#pragma unroll
    for (int nt = 0; nt < 4; ++nt)
#pragma unroll
      for (int r = 0; r < 8; ++r)
        P[(r + 8 * hi) * PS + nt * 16 + rA] = (bf16)sc[nt][r];

    // ---- O += P * V : 2 K-chunks (32 keys) x 8 N-tiles (16 of d) ----
#pragma unroll
    for (int c2 = 0; c2 < 2; ++c2) {
      Frag16 ap;
      ap.h[0] = *(const v8bf*)&P[rA * PS + c2 * 32 + kb8];
      ap.h[1] = *(const v8bf*)&P[rA * PS + c2 * 32 + kb8 + 16];
      const bf16* vp = V + ((size_t)bh * SEQ + kbase + c2 * 32) * HEAD_D;
#pragma unroll
      for (int n = 0; n < 8; ++n) {
        Frag16 bv;  // col = d (lane&15), elements = keys in table order
#pragma unroll
        for (int s = 0; s < 16; ++s)
          bv.v[s] = vp[(s < 8 ? kb8 + s : 8 + kb8 + s) * HEAD_D + n * 16 + rA];
        o[n] = wmma_bf16(ap.v, bv.v, o[n]);
      }
    }
  }

  // ---- normalize and store to attn[(b*SEQ+s), h*HEAD_D + d] ----
  const int b_ = bh >> 5, h_ = bh & (NHEADS - 1);
#pragma unroll
  for (int r = 0; r < 8; ++r) {
    const float linv = 1.0f / l[r];
    const int   srow = q0 + r + 8 * hi;
    const size_t base = ((size_t)b_ * SEQ + srow) * H_DIM + h_ * HEAD_D;
#pragma unroll
    for (int n = 0; n < 8; ++n)
      attn[base + n * 16 + rA] = (bf16)(o[n][r] * linv);
  }
}

// ---------------------------------------------------------------------------
// Launcher
// ---------------------------------------------------------------------------
extern "C" void kernel_launch(void* const* d_in, const int* in_sizes, int n_in,
                              void* d_out, int out_size, void* d_ws,
                              size_t ws_size, hipStream_t stream) {
  const int*   positions = (const int*)d_in[0];
  const float* hidden    = (const float*)d_in[1];
  const float* w_qkv     = (const float*)d_in[2];
  const float* w_o       = (const float*)d_in[3];
  float*       out       = (float*)d_out;

  char* ws = (char*)d_ws;
  size_t off = 0;
  auto balloc = [&](size_t elems) {
    bf16* p = (bf16*)(ws + off);
    off += (elems * sizeof(bf16) + 255) & ~(size_t)255;
    return p;
  };
  bf16* hsb   = balloc((size_t)M_TOT * H_DIM);
  bf16* wqkvT = balloc((size_t)H_DIM * N_QKV);
  bf16* woT   = balloc((size_t)H_DIM * H_DIM);
  bf16* qkv   = balloc((size_t)M_TOT * N_QKV);
  bf16* Qb    = balloc((size_t)BATCH * NHEADS * SEQ * HEAD_D);
  bf16* KTb   = balloc((size_t)BATCH * NHEADS * SEQ * HEAD_D);
  bf16* Vb    = balloc((size_t)BATCH * NHEADS * SEQ * HEAD_D);
  bf16* attn  = balloc((size_t)M_TOT * H_DIM);

  // 1) precision conversion / weight transposes
  {
    const int n4 = (M_TOT * H_DIM) / 4;
    cvt_f32_bf16_k<<<(n4 + 255) / 256, 256, 0, stream>>>(
        (const float4*)hidden, (bf16x4*)hsb, n4);
  }
  transpose_cvt_k<<<dim3(H_DIM / 32, N_QKV / 32), dim3(32, 8), 0, stream>>>(
      w_qkv, wqkvT, N_QKV, H_DIM);
  transpose_cvt_k<<<dim3(H_DIM / 32, H_DIM / 32), dim3(32, 8), 0, stream>>>(
      w_o, woT, H_DIM, H_DIM);

  // 2) QKV projection (clip + bf16 epilogue), TDM-fed
  gemm_bf16_k<1><<<dim3(N_QKV / 128, M_TOT / 128), 256, 0, stream>>>(
      hsb, wqkvT, qkv, M_TOT, N_QKV, H_DIM);

  // 3) RoPE + attention-friendly relayout
  {
    const int nthreads = BATCH * SEQ * NHEADS * 64;  // 2^23
    rope_relayout_k<<<nthreads / 256, 256, 0, stream>>>(
        qkv, positions, Qb, KTb, Vb);
  }

  // 4) causal flash attention
  flash_attn_k<<<dim3(SEQ / 64, BATCH * NHEADS), 128, 0, stream>>>(
      Qb, KTb, Vb, attn);

  // 5) output projection (f32 epilogue -> d_out), TDM-fed
  gemm_bf16_k<0><<<dim3(H_DIM / 128, M_TOT / 128), 256, 0, stream>>>(
      attn, woT, out, M_TOT, H_DIM, H_DIM);
}
